// MultiHeadSelfAttention_85564338471170
// MI455X (gfx1250) — matrix operand order repaired and checked by reading, not compiled, not run on hardware
//
#include <hip/hip_runtime.h>

// ---------------------------------------------------------------------------
// MHSA forward for MI455X (gfx1250, wave32, WMMA).
// B=2, S=4096, D=512, H=8, hd=64.  bf16 WMMA compute, f32 accumulation,
// flash-attention (no [S,S] scores -> ~50MB working set, L2-resident).
// K/V and GEMM tiles staged to LDS via GLOBAL_LOAD_ASYNC_TO_LDS_B128.
// ---------------------------------------------------------------------------

typedef __attribute__((ext_vector_type(16))) __bf16 v16bf;
typedef __attribute__((ext_vector_type(8)))  __bf16 v8bf;
typedef __attribute__((ext_vector_type(8)))  float  v8f;

union Frag { v16bf v; v8bf h[2]; };

#define BB  2
#define SS  4096
#define DD  512
#define HH  8
#define HD  64

__device__ __forceinline__ unsigned short f2bf(float f) {
  unsigned u = __builtin_bit_cast(unsigned, f);
  u += 0x7FFFu + ((u >> 16) & 1u);          // round-to-nearest-even
  return (unsigned short)(u >> 16);
}

__device__ __forceinline__ v8f wmma_bf16(const Frag& a, const Frag& b, v8f c) {
  return __builtin_amdgcn_wmma_f32_16x16x32_bf16(
      false, a.v, false, b.v, (short)0, c, false, false);
}

// LDS offset of a generic pointer into __shared__: flat LDS aperture keeps the
// wave-relative LDS byte offset in addr[31:0] (ISA 07_vmem 10.2).
__device__ __forceinline__ unsigned lds_off(const void* p) {
  return (unsigned)(unsigned long long)p;
}

// Async copy global -> LDS, 16 bytes per lane.  Tracked with ASYNCcnt.
__device__ __forceinline__ void async_b128(unsigned lds, const void* gptr) {
  asm volatile("global_load_async_to_lds_b128 %0, %1, off"
               :: "v"(lds), "v"(gptr) : "memory");
}
__device__ __forceinline__ void wait_async0() {
  asm volatile("s_wait_asynccnt 0x0" ::: "memory");
}

// ---------------------------------------------------------------------------
// f32 -> bf16 elementwise convert (X)
// ---------------------------------------------------------------------------
__global__ void cvt_x_kernel(const float* __restrict__ x,
                             unsigned short* __restrict__ out, int n) {
  int i = blockIdx.x * blockDim.x + threadIdx.x;
  if (i < n) out[i] = f2bf(x[i]);
}

// ---------------------------------------------------------------------------
// Transpose-convert weights: WT[n][k] = W[k][n]  (bf16), one weight per blockIdx.z
// ---------------------------------------------------------------------------
__global__ void cvt_wT_kernel(const float* __restrict__ w0, const float* __restrict__ w1,
                              const float* __restrict__ w2, const float* __restrict__ w3,
                              unsigned short* __restrict__ o0, unsigned short* __restrict__ o1,
                              unsigned short* __restrict__ o2, unsigned short* __restrict__ o3) {
  const float* w; unsigned short* o;
  switch (blockIdx.z) {
    case 0:  w = w0; o = o0; break;
    case 1:  w = w1; o = o1; break;
    case 2:  w = w2; o = o2; break;
    default: w = w3; o = o3; break;
  }
  int i = blockIdx.x * blockDim.x + threadIdx.x;   // 0 .. 512*512-1
  int n = i >> 9, k = i & 511;
  o[n * DD + k] = f2bf(w[k * DD + n]);
}

// ---------------------------------------------------------------------------
// Tiled WMMA GEMM:  out[8192 x 512] = A[8192 x 512] @ W + bias
//   A: bf16 row-major.  WT: bf16 [n][k] (pre-transposed).
//   MODE 0: Q -> bf16 [B,H,S,hd]   MODE 1: K -> bf16 [B,H,S,hd]
//   MODE 2: V -> bf16 [B,H,hd,S]   MODE 3: f32 [B,S,D]
// WG = 256 threads = 8 waves; WG tile 128x64; wave tile 16x64; K-step 32.
// Tiles staged via async-to-LDS.
// ---------------------------------------------------------------------------
template <int MODE>
__global__ __launch_bounds__(256)
void gemm_kernel(const unsigned short* __restrict__ A,
                 const unsigned short* __restrict__ WT,
                 const float* __restrict__ bias,
                 void* __restrict__ outp) {
  __shared__ unsigned short Abuf[128][32];   // [m][k]  8 KB
  __shared__ unsigned short Bbuf[64][32];    // [n][k]  4 KB
  const int t    = threadIdx.x;
  const int lane = t & 31, wave = t >> 5;
  const int tileM = blockIdx.x * 128, tileN = blockIdx.y * 64;
  const int off8  = (lane < 16) ? 0 : 8;
  const int off16 = (lane < 16) ? 0 : 16;
  const int lc    = lane & 15;

  v8f acc[4] = {};

  for (int kk = 0; kk < DD; kk += 32) {
    {   // async-stage A 128x32 (two b128/thread) and B [n][k] 64x32 (one b128/thread)
      int r  = t >> 2, c8 = (t & 3) << 3;
      async_b128(lds_off(&Abuf[r][c8]),      &A[(tileM + r) * DD + kk + c8]);
      async_b128(lds_off(&Abuf[r + 64][c8]), &A[(tileM + r + 64) * DD + kk + c8]);
      async_b128(lds_off(&Bbuf[r][c8]),      &WT[(tileN + r) * DD + kk + c8]);
    }
    wait_async0();
    __syncthreads();

    Frag af;   // A 16x32 fragment
    {
      int row = wave * 16 + lc;
      af.h[0] = *(const v8bf*)&Abuf[row][off8];
      af.h[1] = *(const v8bf*)&Abuf[row][16 + off8];
    }
#pragma unroll
    for (int nt = 0; nt < 4; ++nt) {
      Frag bf;  // B 32x16 fragment: lane = column, K at off8 and 16 + off8
      int col = nt * 16 + lc;
      bf.h[0] = *(const v8bf*)&Bbuf[col][off8];
      bf.h[1] = *(const v8bf*)&Bbuf[col][16 + off8];
      acc[nt] = wmma_bf16(af, bf, acc[nt]);
    }
    __syncthreads();
  }

  // Epilogue: C layout -> VGPR r holds row (r + off8), column = lane&15
#pragma unroll
  for (int nt = 0; nt < 4; ++nt) {
#pragma unroll
    for (int r = 0; r < 8; ++r) {
      int gm = tileM + wave * 16 + r + off8;   // global row  (b*S + s)
      int gn = tileN + nt * 16 + lc;           // global col  (h*hd + d)
      float v = acc[nt][r] + bias[gn];
      if (MODE == 3) {
        ((float*)outp)[gm * DD + gn] = v;
      } else {
        int b = gm >> 12, s = gm & (SS - 1);
        int h = gn >> 6,  d = gn & (HD - 1);
        unsigned short* o = (unsigned short*)outp;
        if (MODE == 2) o[((b * HH + h) * HD + d) * SS + s] = f2bf(v);   // V^T
        else           o[((b * HH + h) * SS + s) * HD + d] = f2bf(v);   // Q / K
      }
    }
  }
}

// ---------------------------------------------------------------------------
// Flash attention: grid (S/128, B*H), 256 threads = 8 waves.
// Each wave owns 16 query rows.  Key blocks of 64: K/V blocks async-staged to
// LDS once per WG (8x reuse across waves).  8 QK^T + 8 PV WMMAs per block.
// Softmax in log2 domain: one v_mul + one v_exp per element.
// ---------------------------------------------------------------------------
__global__ __launch_bounds__(256)
void attn_kernel(const unsigned short* __restrict__ Qb,
                 const unsigned short* __restrict__ Kb,
                 const unsigned short* __restrict__ Vt,
                 unsigned short* __restrict__ attn) {
  __shared__ unsigned short Kbuf[64][64];    // [key][d]   8 KB
  __shared__ unsigned short Vbuf[64][64];    // [d][key]   8 KB
  __shared__ unsigned short Pl[8][16][64];   // per-wave P (C->A layout) 16 KB
  const int t    = threadIdx.x;
  const int lane = t & 31, wave = t >> 5;
  const int off8  = (lane < 16) ? 0 : 8;
  const int off16 = (lane < 16) ? 0 : 16;
  const int lc    = lane & 15;
  const int bh = blockIdx.y;
  const int q0 = blockIdx.x * 128 + wave * 16;
  const float zscale = 0.125f * 1.44269504f;   // (1/sqrt(hd)) * log2(e)

  const unsigned short* Qp  = Qb + (bh * SS + q0) * HD;
  const unsigned short* Kbh = Kb + bh * SS * HD;
  const unsigned short* Vbh = Vt + bh * HD * SS;

  // Q fragments (A matrix, 16x32 over d), held for the whole kernel
  Frag qf[2];
#pragma unroll
  for (int c = 0; c < 2; ++c) {
    int dbase = c * 32 + off8;
    qf[c].h[0] = *(const v8bf*)&Qp[lc * HD + dbase];
    qf[c].h[1] = *(const v8bf*)&Qp[lc * HD + dbase + 16];
  }

  v8f   o[4] = {};
  float mrow[8], lrow[8];
#pragma unroll
  for (int r = 0; r < 8; ++r) { mrow[r] = -1.0e30f; lrow[r] = 0.0f; }

  // staging map: 512 b128-chunks per 64x64 tile; thread covers idx t and t+256
  const int r0 = t >> 3,          c0 = (t & 7) << 3;
  const int r1 = (t + 256) >> 3,  c1 = (t & 7) << 3;

  for (int kb = 0; kb < SS; kb += 64) {
    // ---- async-stage K block [key][d] and V^T block [d][key]
    async_b128(lds_off(&Kbuf[r0][c0]), &Kbh[(kb + r0) * HD + c0]);
    async_b128(lds_off(&Kbuf[r1][c1]), &Kbh[(kb + r1) * HD + c1]);
    async_b128(lds_off(&Vbuf[r0][c0]), &Vbh[r0 * SS + kb + c0]);
    async_b128(lds_off(&Vbuf[r1][c1]), &Vbh[r1 * SS + kb + c1]);
    wait_async0();
    __syncthreads();

    // ---- scores: four 16x16 tiles covering keys [kb, kb+64)
    v8f s[4] = {};
#pragma unroll
    for (int kt = 0; kt < 4; ++kt) {
#pragma unroll
      for (int c = 0; c < 2; ++c) {
        Frag kf;
        int key   = kt * 16 + lc;
        int dbase = c * 32 + off8;
        kf.h[0] = *(const v8bf*)&Kbuf[key][dbase];
        kf.h[1] = *(const v8bf*)&Kbuf[key][dbase + 16];
        s[kt] = wmma_bf16(qf[c], kf, s[kt]);
      }
    }

    // ---- online softmax (log2 domain; stats per lane-half, per reg r)
    float p[4][8], alpha[8];
#pragma unroll
    for (int r = 0; r < 8; ++r) {
      float z0 = s[0][r] * zscale, z1 = s[1][r] * zscale;
      float z2 = s[2][r] * zscale, z3 = s[3][r] * zscale;
      float mr = fmaxf(fmaxf(z0, z1), fmaxf(z2, z3));
      mr = fmaxf(mr, __shfl_xor(mr, 8, 32));
      mr = fmaxf(mr, __shfl_xor(mr, 4, 32));
      mr = fmaxf(mr, __shfl_xor(mr, 2, 32));
      mr = fmaxf(mr, __shfl_xor(mr, 1, 32));
      float mnew = fmaxf(mrow[r], mr);
      alpha[r] = __builtin_amdgcn_exp2f(mrow[r] - mnew);
      mrow[r]  = mnew;
      float e0 = __builtin_amdgcn_exp2f(z0 - mnew);
      float e1 = __builtin_amdgcn_exp2f(z1 - mnew);
      float e2 = __builtin_amdgcn_exp2f(z2 - mnew);
      float e3 = __builtin_amdgcn_exp2f(z3 - mnew);
      p[0][r] = e0; p[1][r] = e1; p[2][r] = e2; p[3][r] = e3;
      float ps = (e0 + e1) + (e2 + e3);
      ps += __shfl_xor(ps, 8, 32);
      ps += __shfl_xor(ps, 4, 32);
      ps += __shfl_xor(ps, 2, 32);
      ps += __shfl_xor(ps, 1, 32);
      lrow[r] = lrow[r] * alpha[r] + ps;
    }
#pragma unroll
    for (int dt = 0; dt < 4; ++dt)
#pragma unroll
      for (int r = 0; r < 8; ++r) o[dt][r] *= alpha[r];

    // ---- P: C-layout regs -> A-layout fragments via wave-private LDS
#pragma unroll
    for (int kt = 0; kt < 4; ++kt)
#pragma unroll
      for (int r = 0; r < 8; ++r)
        Pl[wave][r + off8][kt * 16 + lc] = f2bf(p[kt][r]);
    __builtin_amdgcn_wave_barrier();   // same-wave DS is in-order; stop reordering
    Frag pf[2];
#pragma unroll
    for (int c = 0; c < 2; ++c) {
      pf[c].h[0] = *(const v8bf*)&Pl[wave][lc][c * 32 + off8];
      pf[c].h[1] = *(const v8bf*)&Pl[wave][lc][c * 32 + 16 + off8];
    }

    // ---- O += P @ V   (Vbuf [d][key] gives contiguous B-fragments)
#pragma unroll
    for (int dt = 0; dt < 4; ++dt) {
#pragma unroll
      for (int c = 0; c < 2; ++c) {
        Frag vf;
        int d = dt * 16 + lc;
        vf.h[0] = *(const v8bf*)&Vbuf[d][c * 32 + off8];
        vf.h[1] = *(const v8bf*)&Vbuf[d][c * 32 + 16 + off8];
        o[dt] = wmma_bf16(pf[c], vf, o[dt]);
      }
    }
    __syncthreads();   // before next block overwrites Kbuf/Vbuf
  }

  // ---- normalize + store bf16 [B, S, H*hd]
  float inv[8];
#pragma unroll
  for (int r = 0; r < 8; ++r) inv[r] = 1.0f / lrow[r];
  int b = bh >> 3, h = bh & 7;
#pragma unroll
  for (int dt = 0; dt < 4; ++dt) {
#pragma unroll
    for (int r = 0; r < 8; ++r) {
      int q   = q0 + r + off8;
      int col = h * HD + dt * 16 + lc;
      attn[(b * SS + q) * DD + col] = f2bf(o[dt][r] * inv[r]);
    }
  }
}

// ---------------------------------------------------------------------------
// Host-side launcher.  Workspace layout (needs 42 MiB of d_ws):
//   [0,8M)   Xbf bf16 [8192,512]      [8M,10M)  WqT/WkT/WvT/WoT bf16 [512,512]
//   [10,18M) Q bf16 [B,H,S,hd]        [18,26M)  K bf16 [B,H,S,hd]
//   [26,34M) V^T bf16 [B,H,hd,S]      [34,42M)  attn bf16 [B,S,512]
// ---------------------------------------------------------------------------
extern "C" void kernel_launch(void* const* d_in, const int* in_sizes, int n_in,
                              void* d_out, int out_size, void* d_ws, size_t ws_size,
                              hipStream_t stream) {
  (void)in_sizes; (void)n_in; (void)out_size; (void)ws_size;
  const float* x  = (const float*)d_in[0];
  const float* Wq = (const float*)d_in[1];
  const float* bq = (const float*)d_in[2];
  const float* Wk = (const float*)d_in[3];
  const float* bk = (const float*)d_in[4];
  const float* Wv = (const float*)d_in[5];
  const float* bv = (const float*)d_in[6];
  const float* Wo = (const float*)d_in[7];
  const float* bo = (const float*)d_in[8];

  char* ws = (char*)d_ws;
  unsigned short* Xbf = (unsigned short*)(ws);
  unsigned short* WqT = (unsigned short*)(ws + (8u  << 20));
  unsigned short* WkT = WqT + DD * DD;
  unsigned short* WvT = WkT + DD * DD;
  unsigned short* WoT = WvT + DD * DD;
  unsigned short* Qb  = (unsigned short*)(ws + (10u << 20));
  unsigned short* Kb  = (unsigned short*)(ws + (18u << 20));
  unsigned short* Vt  = (unsigned short*)(ws + (26u << 20));
  unsigned short* At  = (unsigned short*)(ws + (34u << 20));

  const int NX = BB * SS * DD;
  cvt_x_kernel<<<NX / 256, 256, 0, stream>>>(x, Xbf, NX);
  cvt_wT_kernel<<<dim3((DD * DD) / 256, 1, 4), 256, 0, stream>>>(
      Wq, Wk, Wv, Wo, WqT, WkT, WvT, WoT);

  gemm_kernel<0><<<dim3(64, 8), 256, 0, stream>>>(Xbf, WqT, bq, Qb);
  gemm_kernel<1><<<dim3(64, 8), 256, 0, stream>>>(Xbf, WkT, bk, Kb);
  gemm_kernel<2><<<dim3(64, 8), 256, 0, stream>>>(Xbf, WvT, bv, Vt);

  attn_kernel<<<dim3(SS / 128, BB * HH), 256, 0, stream>>>(Qb, Kb, Vt, At);

  gemm_kernel<3><<<dim3(64, 8), 256, 0, stream>>>(At, WoT, bo, d_out);
}
